// ExpanderScatterLinear_8418135900214
// MI455X (gfx1250) — matrix operand order repaired and checked by reading, not compiled, not run on hardware
//
#include <hip/hip_runtime.h>

#define INDIM  2048
#define OUTDIM 2048
#define BATCH  512

typedef __attribute__((ext_vector_type(16))) _Float16 v16h;
typedef __attribute__((ext_vector_type(8)))  float    v8f;

// ---------------------------------------------------------------------------
// 1) zero the dense f32 weight accumulation buffer
// ---------------------------------------------------------------------------
__global__ void zero_f32_kernel(float* __restrict__ p, int n) {
    int i = blockIdx.x * blockDim.x + threadIdx.x;
    if (i < n) p[i] = 0.0f;
}

// ---------------------------------------------------------------------------
// 2) f32 -> f16 conversion (used for x and for the densified W)
// ---------------------------------------------------------------------------
__global__ void cvt_f32_to_f16_kernel(const float* __restrict__ s,
                                      _Float16* __restrict__ d, int n) {
    int i = blockIdx.x * blockDim.x + threadIdx.x;
    if (i < n) d[i] = (_Float16)s[i];
}

// ---------------------------------------------------------------------------
// 3) scatter edges into dense W:  W[o][i] += w   (duplicates -> atomicAdd)
// ---------------------------------------------------------------------------
__global__ void scatter_edges_kernel(const float* __restrict__ w,
                                     const int* __restrict__ ind_in,
                                     const int* __restrict__ ind_out,
                                     float* __restrict__ Wf, int nnz) {
    int e = blockIdx.x * blockDim.x + threadIdx.x;
    if (e < nnz) {
        atomicAdd(&Wf[(size_t)ind_out[e] * INDIM + ind_in[e]], w[e]);
    }
}

// ---------------------------------------------------------------------------
// 4) dense GEMM via CDNA5 WMMA:  out[b][o] = sum_i xh[b][i]*Wh[o][i] + bias[o]
//    Register-blocked: one wave computes a 32(batch) x 64(outcol) f32 tile
//    = 2 M-subtiles x 4 N-subtiles = 8 accumulators. Per K=32 step:
//    2 A-tile + 4 B-tile operand fetches feed 8 WMMAs (1.5 b128 loads/WMMA).
// ---------------------------------------------------------------------------
__global__ void gemm_wmma_kernel(const _Float16* __restrict__ xh,
                                 const _Float16* __restrict__ wh,
                                 const float* __restrict__ bias,
                                 float* __restrict__ out) {
    const int lane = threadIdx.x;                           // 0..31
    const int r    = lane & 15;                             // row within subtile
    const int h    = lane >> 4;                             // lane half (K-half select)
    const int nt   = blockIdx.x * blockDim.y + threadIdx.y; // 0..31, 64-col blocks
    const int mt   = blockIdx.y;                            // 0..15, 32-row blocks

    // A rows: batch index; B rows: output column index (both K-contiguous in mem)
    const _Float16* ap0 = xh + (size_t)(mt * 32 + r) * INDIM;
    const _Float16* ap1 = ap0 + (size_t)16 * INDIM;
    const _Float16* bp0 = wh + (size_t)(nt * 64 + r) * INDIM;

    v8f acc[2][4] = {};

    for (int kb = 0; kb < INDIM; kb += 32) {
        v16h a[2], b[4];
        // CDNA5 16-bit A 16x32 layout: lanes 0-15 hold K={0..7,16..23},
        // lanes 16-31 hold K={8..15,24..31}; 2 halfs per VGPR.
        // CDNA5 16-bit B 32x16 layout: lanes 0-15 hold K={0..7,16..23},
        // lanes 16-31 hold K={8..15,24..31}; 2 halfs per VGPR.
        #pragma unroll
        for (int v = 0; v < 8; ++v) {
            int ca = ((((v >> 2) << 3) + (h << 2) + (v & 3)) << 1); // A half index
            int cb = ((((v >> 2) << 3) + (h << 2) + (v & 3)) << 1); // B half index
            a[0][2 * v]     = ap0[kb + ca];
            a[0][2 * v + 1] = ap0[kb + ca + 1];
            a[1][2 * v]     = ap1[kb + ca];
            a[1][2 * v + 1] = ap1[kb + ca + 1];
            #pragma unroll
            for (int s = 0; s < 4; ++s) {
                const _Float16* bp = bp0 + (size_t)(s * 16) * INDIM;
                b[s][2 * v]     = bp[kb + cb];
                b[s][2 * v + 1] = bp[kb + cb + 1];
            }
        }
        #pragma unroll
        for (int m = 0; m < 2; ++m) {
            #pragma unroll
            for (int s = 0; s < 4; ++s) {
                // 8 args: (neg_a, A, neg_b, B, c_mod, C, reuse_a, reuse_b)
                acc[m][s] = __builtin_amdgcn_wmma_f32_16x16x32_f16(
                    false, a[m], false, b[s], (short)0, acc[m][s], false, false);
            }
        }
    }

    // C/D f32 16x16 layout: VGPR rr -> M=rr (lanes 0-15) / M=rr+8 (lanes 16-31),
    // N = lane & 15. Fuse bias add on the way out.
    #pragma unroll
    for (int s = 0; s < 4; ++s) {
        const int   ocol = nt * 64 + s * 16 + r;
        const float bv   = bias[ocol];
        #pragma unroll
        for (int m = 0; m < 2; ++m) {
            #pragma unroll
            for (int rr = 0; rr < 8; ++rr) {
                int brow = mt * 32 + m * 16 + rr + h * 8;
                out[(size_t)brow * OUTDIM + ocol] = acc[m][s][rr] + bv;
            }
        }
    }
}

// ---------------------------------------------------------------------------
// Launch: densify sparse W, convert to f16, then WMMA GEMM with fused bias.
// ws layout: [0,16MB) Wf f32 | [16MB,24MB) Wh f16 | [24MB,26MB) xh f16
// ---------------------------------------------------------------------------
extern "C" void kernel_launch(void* const* d_in, const int* in_sizes, int n_in,
                              void* d_out, int out_size, void* d_ws, size_t ws_size,
                              hipStream_t stream) {
    const float* x       = (const float*)d_in[0];
    const float* weight  = (const float*)d_in[1];
    const float* bias    = (const float*)d_in[2];
    const int*   ind_in  = (const int*)d_in[3];
    const int*   ind_out = (const int*)d_in[4];
    const int    nnz     = in_sizes[1];
    float*       out     = (float*)d_out;

    char* ws = (char*)d_ws;
    float*    Wf = (float*)ws;                                        // 16 MB
    _Float16* Wh = (_Float16*)(ws + (size_t)OUTDIM * INDIM * 4);      //  8 MB
    _Float16* xh = (_Float16*)(ws + (size_t)OUTDIM * INDIM * 6);      //  2 MB

    const int NW = OUTDIM * INDIM;   // 4,194,304
    const int NX = BATCH * INDIM;    // 1,048,576

    zero_f32_kernel<<<(NW + 255) / 256, 256, 0, stream>>>(Wf, NW);
    cvt_f32_to_f16_kernel<<<(NX + 255) / 256, 256, 0, stream>>>(x, xh, NX);
    scatter_edges_kernel<<<(nnz + 255) / 256, 256, 0, stream>>>(weight, ind_in, ind_out, Wf, nnz);
    cvt_f32_to_f16_kernel<<<(NW + 255) / 256, 256, 0, stream>>>(Wf, Wh, NW);

    // 32 col-blocks (64 wide) x 16 batch-blocks (32 rows); 8 waves per block
    gemm_wmma_kernel<<<dim3(4, 16), dim3(32, 8), 0, stream>>>(xh, Wh, bias, out);
}
